// M100_14336600834653
// MI455X (gfx1250) — matrix operand order repaired and checked by reading, not compiled, not run on hardware
//
#include <hip/hip_runtime.h>

#define H 128
#define TMAX 7305   // harness T; LDS forcing-cache capacity (guarded fallback)

typedef __attribute__((ext_vector_type(16))) _Float16 v16h;
typedef __attribute__((ext_vector_type(8)))  float    v8f;

__device__ __forceinline__ float stepf(float v) {
    return 0.5f * (tanhf(5.0f * v) + 1.0f);
}

// jnp.interp on a uniform grid t_grid = arange(T)
__device__ __forceinline__ float interp1(const float* __restrict__ d, int T, float t) {
    if (t <= 0.0f) return d[0];
    if (t >= (float)(T - 1)) return d[T - 1];
    int i = (int)t;
    if (i > T - 2) i = T - 2;
    float f = t - (float)i;
    return fmaf(f, d[i + 1] - d[i], d[i]);
}

// ---------------------------------------------------------------------------
// Phase 1: strictly sequential RK4 scan. One workgroup (4 waves). All weights
// AND all three forcing time-series live in LDS (~159 KB of the 320 KB WGP
// pool), so the serial dependence chain never touches global memory except
// the software-pipelined t_eval prefetch and the trajectory stores.
// ---------------------------------------------------------------------------
__global__ __launch_bounds__(H) void hydro_scan(
    const float* __restrict__ x, const float* __restrict__ t_eval,
    const float* __restrict__ precp, const float* __restrict__ temp,
    const float* __restrict__ lday,
    const float* __restrict__ W1, const float* __restrict__ b1,
    const float* __restrict__ W2, const float* __restrict__ b2,
    const float* __restrict__ W3, const float* __restrict__ b3,
    float* __restrict__ sol, int T)
{
    __shared__ float sW1[4 * H];
    __shared__ float sb1[H];
    __shared__ float sW2[H * H];      // 64 KB, f32: ODE state accuracy matters
    __shared__ float sb2[H];
    __shared__ float sW3[H * 5];
    __shared__ float sb3[5];
    __shared__ float sh1[H];
    __shared__ float sS[2];
    __shared__ float sred[4][5];
    __shared__ float sPr[TMAX];       // forcing caches: 3 x 28.5 KB
    __shared__ float sTe[TMAX];
    __shared__ float sLd[TMAX];

    const int tid = threadIdx.x;

    for (int i = tid; i < 4 * H; i += H) sW1[i] = W1[i];
    for (int i = tid; i < H * H; i += H) sW2[i] = W2[i];
    for (int i = tid; i < 5 * H; i += H) sW3[i] = W3[i];
    sb1[tid] = b1[tid];
    sb2[tid] = b2[tid];
    if (tid < 5) sb3[tid] = b3[tid];

    const bool useLds = (T <= TMAX);
    if (useLds) {
        for (int i = tid; i < T; i += H) {
            sPr[i] = precp[i];
            sTe[i] = temp[i];
            sLd[i] = lday[i];
        }
    }
    if (tid == 0) {
        sS[0] = x[0]; sS[1] = x[1];       // y0 = x[0,:2]
        sol[0] = x[0]; sol[1] = x[1];
    }
    __syncthreads();

    float Sb0 = sS[0], Sb1 = sS[1];       // meaningful on thread 0 only
    float ka0 = 0.0f, ka1 = 0.0f;

    // software-pipelined t_eval: loads issued one full RK4 step early
    float tcur  = t_eval[0];
    float tnext = (T > 1) ? t_eval[1] : tcur;

    for (int n = 0; n < T - 1; ++n) {
        const int np2 = (n + 2 < T) ? n + 2 : T - 1;
        const float tpref = t_eval[np2];      // consumed at loop bottom
        const float t0 = tcur;
        const float hh = tnext - tcur;
        for (int st = 0; st < 4; ++st) {
            float ts = t0 + ((st == 0) ? 0.0f : (st == 3) ? hh : 0.5f * hh);
            float pr, te;
            if (useLds) {
                pr = interp1(sPr, T, ts);
                te = interp1(sTe, T, ts);
            } else {
                pr = interp1(precp, T, ts);
                te = interp1(temp, T, ts);
            }
            float z0 = sS[0], z1 = sS[1];

            // layer 1: each thread one hidden unit
            float a1 = sb1[tid];
            a1 = fmaf(z0, sW1[0 * H + tid], a1);
            a1 = fmaf(z1, sW1[1 * H + tid], a1);
            a1 = fmaf(pr, sW1[2 * H + tid], a1);
            a1 = fmaf(te, sW1[3 * H + tid], a1);
            sh1[tid] = tanhf(a1);
            __syncthreads();

            // layer 2: broadcast h1 from LDS, conflict-free W2 column reads
            float a2 = sb2[tid];
            #pragma unroll 8
            for (int i = 0; i < H; ++i) a2 = fmaf(sh1[i], sW2[i * H + tid], a2);
            float h2v = tanhf(a2);

            // layer 3: 5 outputs via wave32 xor-reduction
            float p0 = h2v * sW3[tid * 5 + 0];
            float p1 = h2v * sW3[tid * 5 + 1];
            float p2 = h2v * sW3[tid * 5 + 2];
            float p3 = h2v * sW3[tid * 5 + 3];
            float p4 = h2v * sW3[tid * 5 + 4];
            #pragma unroll
            for (int m = 16; m >= 1; m >>= 1) {
                p0 += __shfl_xor(p0, m, 32);
                p1 += __shfl_xor(p1, m, 32);
                p2 += __shfl_xor(p2, m, 32);
                p3 += __shfl_xor(p3, m, 32);
                p4 += __shfl_xor(p4, m, 32);
            }
            if ((tid & 31) == 0) {
                int w = tid >> 5;
                sred[w][0] = p0; sred[w][1] = p1; sred[w][2] = p2;
                sred[w][3] = p3; sred[w][4] = p4;
            }
            __syncthreads();

            if (tid == 0) {
                float o0 = sb3[0] + sred[0][0] + sred[1][0] + sred[2][0] + sred[3][0];
                float o1 = sb3[1] + sred[0][1] + sred[1][1] + sred[2][1] + sred[3][1];
                float o2 = sb3[2] + sred[0][2] + sred[1][2] + sred[2][2] + sred[3][2];
                float o3 = sb3[3] + sred[0][3] + sred[1][3] + sred[2][3] + sred[3][3];
                float o4 = sb3[4] + sred[0][4] + sred[1][4] + sred[2][4] + sred[3][4];
                float ld;
                if (useLds) ld = interp1(sLd, T, ts);
                else        ld = interp1(lday, T, ts);
                float melt = fmaxf(stepf(z0), 0.0f) * sinhf(o2);
                float d1 = fmaxf(sinhf(o3) * stepf(-te), 0.0f) - melt;
                float s1s = stepf(z1);
                float d2 = fmaxf(sinhf(o4), 0.0f) + melt
                           - s1s * ld * expf(o0) - s1s * expf(o1);
                if (st == 0) {
                    ka0 = d1; ka1 = d2;
                    sS[0] = fmaf(0.5f * hh, d1, Sb0); sS[1] = fmaf(0.5f * hh, d2, Sb1);
                } else if (st == 1) {
                    ka0 += 2.0f * d1; ka1 += 2.0f * d2;
                    sS[0] = fmaf(0.5f * hh, d1, Sb0); sS[1] = fmaf(0.5f * hh, d2, Sb1);
                } else if (st == 2) {
                    ka0 += 2.0f * d1; ka1 += 2.0f * d2;
                    sS[0] = fmaf(hh, d1, Sb0); sS[1] = fmaf(hh, d2, Sb1);
                } else {
                    ka0 += d1; ka1 += d2;
                    Sb0 = fmaf(hh * (1.0f / 6.0f), ka0, Sb0);
                    Sb1 = fmaf(hh * (1.0f / 6.0f), ka1, Sb1);
                    sS[0] = Sb0; sS[1] = Sb1;
                    sol[2 * (n + 1)]     = Sb0;
                    sol[2 * (n + 1) + 1] = Sb1;
                }
            }
            __syncthreads();
        }
        tcur = tnext;
        tnext = tpref;
    }
}

// ---------------------------------------------------------------------------
// Phase 2: batched MLP head over all T rows using f16 WMMA (f32 accumulate).
// Only o[:,1] is needed. Each wave computes a 16-row x 128-col strip:
//   32 x v_wmma_f32_16x16x32_f16 per wave, A built in-register per the
//   contiguous-k A layout (k = 16*half + s), B (W2^T f16) staged in LDS in the same k order
//   (one contiguous aligned 32B load per fragment).
// ---------------------------------------------------------------------------
__global__ __launch_bounds__(256) void hydro_head(
    const float* __restrict__ x, const float* __restrict__ sol,
    const float* __restrict__ W1, const float* __restrict__ b1,
    const float* __restrict__ W2, const float* __restrict__ b2,
    const float* __restrict__ W3, const float* __restrict__ b3,
    float* __restrict__ out, int T)
{
    __shared__ __attribute__((aligned(32))) _Float16 sW2t[H * H]; // [n][k] = W2[k][n]
    __shared__ float sW1[4 * H];
    __shared__ float sb1[H];
    __shared__ float sb2[H];
    __shared__ float sw3[H];

    const int tid = threadIdx.x;
    for (int e = tid; e < H * H; e += 256) {
        int nn = e >> 7, kk = e & (H - 1);
        sW2t[e] = (_Float16)W2[kk * H + nn];
    }
    for (int e = tid; e < 4 * H; e += 256) sW1[e] = W1[e];
    if (tid < H) {
        sb1[tid] = b1[tid];
        sb2[tid] = b2[tid];
        sw3[tid] = W3[tid * 5 + 1];   // only output column 1 survives
    }
    __syncthreads();

    const int lane = tid & 31;
    const int wave = tid >> 5;
    const int m0   = (blockIdx.x * 8 + wave) * 16;
    const int half = lane >> 4;          // A/B lane-half (K interleave)
    const int l15  = lane & 15;

    // this lane owns A-row m0+l15 (both halves hold the same M, different K)
    int r = m0 + l15; if (r > T - 1) r = T - 1;
    const float z0 = sol[2 * r], z1 = sol[2 * r + 1];
    const float z2 = x[4 * r + 2], z3 = x[4 * r + 3];

    v8f acc[8] = {};

    const int koff = half * 8;           // A layout: lanes>=16 hold K+16 window
    #pragma unroll
    for (int s = 0; s < 4; ++s) {
        // Build A fragment in registers: layer1 + tanh + f16 pack, in contiguous k order:
        // VGPR v -> K = s*32 + 2v, 2v+1 (+16 for upper lanes)
        v16h afrag;
        #pragma unroll
        for (int v = 0; v < 8; ++v) {
            int kb = s * 32 + 2 * v + 2 * koff;
            float h0 = sb1[kb];
            h0 = fmaf(z0, sW1[kb],         h0);
            h0 = fmaf(z1, sW1[H + kb],     h0);
            h0 = fmaf(z2, sW1[2 * H + kb], h0);
            h0 = fmaf(z3, sW1[3 * H + kb], h0);
            float h1 = sb1[kb + 1];
            h1 = fmaf(z0, sW1[kb + 1],         h1);
            h1 = fmaf(z1, sW1[H + kb + 1],     h1);
            h1 = fmaf(z2, sW1[2 * H + kb + 1], h1);
            h1 = fmaf(z3, sW1[3 * H + kb + 1], h1);
            afrag[2 * v]     = (_Float16)tanhf(h0);
            afrag[2 * v + 1] = (_Float16)tanhf(h1);
        }
        #pragma unroll
        for (int nn = 0; nn < 8; ++nn) {
            // B layout: lanes 0-15 hold K = s*32 + 0..15, lanes 16-31 hold +16..31
            const v16h bfrag =
                *(const v16h*)&sW2t[(nn * 16 + l15) * H + s * 32 + half * 16];
            acc[nn] = __builtin_amdgcn_wmma_f32_16x16x32_f16(
                false, afrag, false, bfrag, (short)0, acc[nn], false, false);
        }
    }

    // Epilogue: h2 = tanh(acc + b2); o1 = h2 . W3[:,1] + b3[1]; out = exp(o1)
    // C layout: vgpr rr -> M = rr (lanes 0-15) or 8+rr (lanes 16-31), N = n*16+l15
    const float b3_1 = b3[1];
    #pragma unroll
    for (int rr = 0; rr < 8; ++rr) {
        float sacc = 0.0f;
        #pragma unroll
        for (int nn = 0; nn < 8; ++nn) {
            int N = nn * 16 + l15;
            sacc += tanhf(acc[nn][rr] + sb2[N]) * sw3[N];
        }
        sacc += __shfl_xor(sacc, 1, 32);
        sacc += __shfl_xor(sacc, 2, 32);
        sacc += __shfl_xor(sacc, 4, 32);
        sacc += __shfl_xor(sacc, 8, 32);
        int row = m0 + rr + half * 8;
        if (l15 == 0 && row < T) out[row] = expf(sacc + b3_1);
    }
}

extern "C" void kernel_launch(void* const* d_in, const int* in_sizes, int n_in,
                              void* d_out, int out_size, void* d_ws, size_t ws_size,
                              hipStream_t stream) {
    const float* x      = (const float*)d_in[0];
    const float* t_eval = (const float*)d_in[1];
    // d_in[2] = t_grid: arange(T), handled implicitly by interp1
    const float* precp  = (const float*)d_in[3];
    const float* temp   = (const float*)d_in[4];
    const float* lday   = (const float*)d_in[5];
    const float* W1 = (const float*)d_in[6];
    const float* b1 = (const float*)d_in[7];
    const float* W2 = (const float*)d_in[8];
    const float* b2 = (const float*)d_in[9];
    const float* W3 = (const float*)d_in[10];
    const float* b3 = (const float*)d_in[11];
    const int T = in_sizes[1];

    float* sol = (float*)d_ws;    // T x 2 trajectory

    hydro_scan<<<1, H, 0, stream>>>(x, t_eval, precp, temp, lday,
                                    W1, b1, W2, b2, W3, b3, sol, T);

    int mtiles = (T + 15) / 16;
    int blocks = (mtiles + 7) / 8;
    hydro_head<<<blocks, 256, 0, stream>>>(x, sol, W1, b1, W2, b2, W3, b3,
                                           (float*)d_out, T);
}
